// DoubleAttention_6253472383446
// MI455X (gfx1250) — matrix operand order repaired and checked by reading, not compiled, not run on hardware
//
#include <hip/hip_runtime.h>
#include <hip/hip_bf16.h>

// ---------------------------------------------------------------------------
// DoubleAttention pipeline for gfx1250 (MI455X), wave32 + WMMA bf16.
//
// Dims (from reference):
//   B=1024, S_OUT=30, S_IN=17, D_IN=3, H=4
//   inner: dkq=6, dv=8, hid=256, out=64   -> fc1 K=136 (pad 160), fc2 K=1024
//   outer: dkq=64, dv=64, hid=1024, out=1080 (pad 1088)
// ---------------------------------------------------------------------------

typedef __attribute__((ext_vector_type(16))) __bf16 v16bf;
typedef __attribute__((ext_vector_type(8)))  float  v8f;

__device__ __forceinline__ unsigned short f32_to_bf16(float f) {
    unsigned u = __float_as_uint(f);
    unsigned r = u + 0x7FFFu + ((u >> 16) & 1u);   // round-to-nearest-even
    return (unsigned short)(r >> 16);
}
__device__ __forceinline__ float bf16_to_f32(unsigned short h) {
    return __uint_as_float(((unsigned)h) << 16);
}

// ---------------------------------------------------------------------------
// bf16 WMMA GEMM:  C[M,N] = epi( A[M,K] @ Wt[N,K]^T + bias[N] )
// A row-major bf16 (ushort), Wt stored TRANSPOSED row-major [N,K] bf16.
// K % 32 == 0, M % 64 == 0, N % 16 == 0.
// One wave computes a 64x16 tile (4 m-subtiles): the B (weight) fragment is
// loaded once per K-step and reused by 4 WMMAs (register blocking).
// 4 waves per block span 64 output columns.
// mode: 0 = f32 out (bias only), 1 = bf16 out (bias + tanh), 2 = bf16 out (bias)
// ---------------------------------------------------------------------------
__global__ __launch_bounds__(128)
void gemm_bf16_wmma(const unsigned short* __restrict__ A,
                    const unsigned short* __restrict__ Wt,
                    const float* __restrict__ bias,
                    void* __restrict__ Cout,
                    int M, int N, int K, int mode)
{
    const int wave  = threadIdx.x >> 5;
    const int lane  = threadIdx.x & 31;
    const int ntile = blockIdx.x * 4 + wave;     // 16 output columns per wave
    const int mbase = blockIdx.y * 64;           // 64 output rows per wave
    if (ntile * 16 >= N) return;

    const int hi = lane >> 4;      // half-wave selector
    const int ml = lane & 15;

    const int bcol = ntile * 16 + ml;            // output column / Wt row

    const unsigned* __restrict__ Br = (const unsigned*)(Wt + (size_t)bcol * K);
    const unsigned* __restrict__ Ar0 = (const unsigned*)(A + (size_t)(mbase + ml) * K);
    const size_t rowStride16 = (size_t)16 * K / 2;   // 16 rows in dwords

    v8f acc0 = {}, acc1 = {}, acc2 = {}, acc3 = {};

    for (int k0 = 0; k0 < K; k0 += 32) {
        union { v16bf v; unsigned u[8]; } bfm, a0, a1, a2, a3;
        // B 32x16: lane = column; lanes 0-15 K={0..7,16..23}, lanes 16-31 K={8..15,24..31}.
        #pragma unroll
        for (int j = 0; j < 8; ++j) {
            int kb = k0 + ((j < 4) ? (2 * j) : (16 + 2 * (j - 4))) + (hi ? 8 : 0);
            bfm.u[j] = Br[kb >> 1];
        }
        // A 16x32 bf16 layout: lanes 0-15 hold K pairs {0,1..6,7 | 16..23},
        // lanes 16-31 hold K pairs {8..15 | 24..31}.
        #pragma unroll
        for (int j = 0; j < 8; ++j) {
            int ka = k0 + ((j < 4) ? (2 * j) : (16 + 2 * (j - 4))) + (hi ? 8 : 0);
            int di = ka >> 1;
            a0.u[j] = Ar0[di];
            a1.u[j] = Ar0[di + rowStride16];
            a2.u[j] = Ar0[di + 2 * rowStride16];
            a3.u[j] = Ar0[di + 3 * rowStride16];
        }
        acc0 = __builtin_amdgcn_wmma_f32_16x16x32_bf16(false, a0.v, false, bfm.v,
                                                       (short)0, acc0, false, false);
        acc1 = __builtin_amdgcn_wmma_f32_16x16x32_bf16(false, a1.v, false, bfm.v,
                                                       (short)0, acc1, false, false);
        acc2 = __builtin_amdgcn_wmma_f32_16x16x32_bf16(false, a2.v, false, bfm.v,
                                                       (short)0, acc2, false, false);
        acc3 = __builtin_amdgcn_wmma_f32_16x16x32_bf16(false, a3.v, false, bfm.v,
                                                       (short)0, acc3, false, false);
    }

    const int   ocol = ntile * 16 + ml;
    const float bv   = bias ? bias[ocol] : 0.0f;

    v8f accs[4] = {acc0, acc1, acc2, acc3};
    #pragma unroll
    for (int s = 0; s < 4; ++s) {
        #pragma unroll
        for (int r = 0; r < 8; ++r) {
            int   orow = mbase + s * 16 + r + 8 * hi;  // C/D: VGPR r -> row r+8*hi
            float v    = accs[s][r] + bv;
            if (mode == 1) v = tanhf(v);
            size_t idx = (size_t)orow * N + ocol;
            if (mode == 0) ((float*)Cout)[idx]          = v;
            else           ((unsigned short*)Cout)[idx] = f32_to_bf16(v);
        }
    }
}

// ---------------------------------------------------------------------------
// Weight pack: Wt[n*Kp + k] = bf16(W[k*N + n]) with zero padding.
// ---------------------------------------------------------------------------
__global__ void pack_wt(const float* __restrict__ W, unsigned short* __restrict__ Wt,
                        int K, int N, int Kp, int Np)
{
    int i = blockIdx.x * blockDim.x + threadIdx.x;
    if (i >= Np * Kp) return;
    int n = i / Kp, k = i - n * Kp;
    Wt[i] = (n < N && k < K) ? f32_to_bf16(W[(size_t)k * N + n]) : (unsigned short)0;
}

// Combined outer q/k/v weights -> Wt[768, 64], bias[768].
// n in [0,256)=Q, [256,512)=K, [512,768)=V; within group h = (n>>6)&3, c = n&63.
__global__ void pack_qkv(const float* __restrict__ owq, const float* __restrict__ owk,
                         const float* __restrict__ owv, const float* __restrict__ obq,
                         const float* __restrict__ obk, const float* __restrict__ obv,
                         unsigned short* __restrict__ Wt, float* __restrict__ bias)
{
    int i = blockIdx.x * blockDim.x + threadIdx.x;   // 768*64
    if (i >= 768 * 64) return;
    int n = i >> 6, k = i & 63;
    int grp = n >> 8;
    int h = (n >> 6) & 3, c = n & 63;
    const float* w = (grp == 0) ? owq : ((grp == 1) ? owk : owv);
    Wt[(size_t)n * 64 + k] = f32_to_bf16(w[((size_t)h * 64 + k) * 64 + c]);
    if (k == 0) {
        const float* bb = (grp == 0) ? obq : ((grp == 1) ? obk : obv);
        bias[n] = bb[h * 64 + c];
    }
}

__global__ void pad_bias(const float* __restrict__ src, float* __restrict__ dst,
                         int n, int np)
{
    int i = blockIdx.x * blockDim.x + threadIdx.x;
    if (i < np) dst[i] = (i < n) ? src[i] : 0.0f;
}

// ---------------------------------------------------------------------------
// Inner attention: one block per (b, s_out). 17 positions, 4 heads, d_in=3,
// dkq=6, dv=8. Emits bf16 A-matrix rows for inner fc1: row m=(bs*4+h),
// cols 0..135 = ctx[p*8+d], cols 136..159 = 0 (K padding).
// ---------------------------------------------------------------------------
__global__ __launch_bounds__(128)
void inner_attn(const float* __restrict__ x,
                const float* __restrict__ iwq, const float* __restrict__ ibq,
                const float* __restrict__ iwk, const float* __restrict__ ibk,
                const float* __restrict__ iwv, const float* __restrict__ ibv,
                unsigned short* __restrict__ ctxiA)
{
    const int bs = blockIdx.x;                 // b*30 + s
    __shared__ float xs[17][3];
    __shared__ float qs[4][17][6];
    __shared__ float ks[4][17][6];
    __shared__ float vs[4][17][8];
    const int t = threadIdx.x;

    if (t < 51) ((float*)xs)[t] = x[(size_t)bs * 51 + t];
    __syncthreads();

    if (t < 68) {
        int h = t / 17, p = t % 17;
        #pragma unroll
        for (int kk = 0; kk < 6; ++kk) {
            float aq = ibq[h * 6 + kk], ak = ibk[h * 6 + kk];
            #pragma unroll
            for (int d = 0; d < 3; ++d) {
                aq += xs[p][d] * iwq[(h * 3 + d) * 6 + kk];
                ak += xs[p][d] * iwk[(h * 3 + d) * 6 + kk];
            }
            qs[h][p][kk] = aq; ks[h][p][kk] = ak;
        }
        #pragma unroll
        for (int kk = 0; kk < 8; ++kk) {
            float av = ibv[h * 8 + kk];
            #pragma unroll
            for (int d = 0; d < 3; ++d) av += xs[p][d] * iwv[(h * 3 + d) * 8 + kk];
            vs[h][p][kk] = av;
        }
    }
    __syncthreads();

    if (t < 68) {
        int h = t / 17, p = t % 17;
        const float inv = 0.40824829046386296f;   // 1/sqrt(6)
        float sc[17], mx = -1e30f;
        #pragma unroll
        for (int j = 0; j < 17; ++j) {
            float s = 0.0f;
            #pragma unroll
            for (int kk = 0; kk < 6; ++kk) s += qs[h][p][kk] * ks[h][j][kk];
            sc[j] = s * inv;
            mx = fmaxf(mx, sc[j]);
        }
        float den = 0.0f;
        #pragma unroll
        for (int j = 0; j < 17; ++j) { sc[j] = __expf(sc[j] - mx); den += sc[j]; }
        float rd = 1.0f / den;
        size_t m = (size_t)bs * 4 + h;
        #pragma unroll
        for (int d = 0; d < 8; ++d) {
            float c = 0.0f;
            #pragma unroll
            for (int j = 0; j < 17; ++j) c += sc[j] * vs[h][j][d];
            ctxiA[m * 160 + p * 8 + d] = f32_to_bf16(c * rd);
        }
    }
    if (t < 96) {     // zero the K-pad columns 136..159 for the 4 head rows
        int h = t / 24, c = 136 + (t % 24);
        ctxiA[((size_t)bs * 4 + h) * 160 + c] = 0;
    }
}

// ---------------------------------------------------------------------------
// Outer attention: one block per (b, h). Q/K/V in LDS, 30x30 softmax, emits
// bf16 rows [b*4+h, 1920] for outer fc1.
// ---------------------------------------------------------------------------
__global__ __launch_bounds__(256)
void outer_attn(const unsigned short* __restrict__ qkv,   // [B*30, 768] bf16
                unsigned short* __restrict__ ctxo)        // [B*4, 1920] bf16
{
    const int b = blockIdx.x >> 2;
    const int h = blockIdx.x & 3;
    __shared__ float Q[30][64], Kk[30][64], V[30][64];
    __shared__ float attn[30][32];
    const int t = threadIdx.x;

    for (int i = t; i < 1920; i += 256) {
        int p = i >> 6, d = i & 63;
        size_t base = ((size_t)(b * 30 + p)) * 768 + h * 64 + d;
        Q[p][d]  = bf16_to_f32(qkv[base]);
        Kk[p][d] = bf16_to_f32(qkv[base + 256]);
        V[p][d]  = bf16_to_f32(qkv[base + 512]);
    }
    __syncthreads();

    if (t < 30) {
        float sc[30], mx = -1e30f;
        for (int j = 0; j < 30; ++j) {
            float s = 0.0f;
            #pragma unroll
            for (int d = 0; d < 64; ++d) s += Q[t][d] * Kk[j][d];
            sc[j] = s * 0.125f;                  // 1/sqrt(64)
            mx = fmaxf(mx, sc[j]);
        }
        float den = 0.0f;
        for (int j = 0; j < 30; ++j) { sc[j] = __expf(sc[j] - mx); den += sc[j]; }
        float rd = 1.0f / den;
        for (int j = 0; j < 30; ++j) attn[t][j] = sc[j] * rd;
    }
    __syncthreads();

    for (int i = t; i < 1920; i += 256) {
        int p = i >> 6, d = i & 63;
        float c = 0.0f;
        #pragma unroll
        for (int j = 0; j < 30; ++j) c += attn[p][j] * V[j][d];
        ctxo[((size_t)(b * 4 + h)) * 1920 + p * 64 + d] = f32_to_bf16(c);
    }
}

// Strip N padding 1088 -> 1080 into d_out.
__global__ void final_copy(const float* __restrict__ C, float* __restrict__ out)
{
    int i = blockIdx.x * blockDim.x + threadIdx.x;
    if (i >= 1024 * 1080) return;
    int b = i / 1080, c = i - b * 1080;
    out[i] = C[(size_t)b * 1088 + c];
}

// ---------------------------------------------------------------------------
extern "C" void kernel_launch(void* const* d_in, const int* in_sizes, int n_in,
                              void* d_out, int out_size, void* d_ws, size_t ws_size,
                              hipStream_t stream)
{
    (void)in_sizes; (void)n_in; (void)out_size; (void)ws_size;
    const float* x   = (const float*)d_in[0];
    const float* iwq = (const float*)d_in[1];  const float* ibq = (const float*)d_in[2];
    const float* iwk = (const float*)d_in[3];  const float* ibk = (const float*)d_in[4];
    const float* iwv = (const float*)d_in[5];  const float* ibv = (const float*)d_in[6];
    const float* iw1 = (const float*)d_in[7];  const float* ib1 = (const float*)d_in[8];
    const float* iw2 = (const float*)d_in[9];  const float* ib2 = (const float*)d_in[10];
    const float* owq = (const float*)d_in[11]; const float* obq = (const float*)d_in[12];
    const float* owk = (const float*)d_in[13]; const float* obk = (const float*)d_in[14];
    const float* owv = (const float*)d_in[15]; const float* obv = (const float*)d_in[16];
    const float* ow1 = (const float*)d_in[17]; const float* ob1 = (const float*)d_in[18];
    const float* ow2 = (const float*)d_in[19]; const float* ob2 = (const float*)d_in[20];

    // ---- workspace layout (256B aligned) ----
    char*  ws  = (char*)d_ws;
    size_t off = 0;
    auto alloc = [&](size_t bytes) -> char* {
        char* p = ws + off;
        off += (bytes + 255) & ~(size_t)255;
        return p;
    };
    unsigned short* iw1t  = (unsigned short*)alloc((size_t)256  * 160  * 2);
    unsigned short* iw2t  = (unsigned short*)alloc((size_t)64   * 1024 * 2);
    unsigned short* qkvWt = (unsigned short*)alloc((size_t)768  * 64   * 2);
    unsigned short* ow1t  = (unsigned short*)alloc((size_t)1024 * 1920 * 2);
    unsigned short* ow2t  = (unsigned short*)alloc((size_t)1088 * 4096 * 2);
    float*          qkvB  = (float*)alloc((size_t)768 * 4);
    float*          ob2p  = (float*)alloc((size_t)1088 * 4);
    unsigned short* ctxiA = (unsigned short*)alloc((size_t)122880 * 160 * 2);
    unsigned short* C1    = (unsigned short*)alloc((size_t)122880 * 256 * 2);
    unsigned short* seqA  = (unsigned short*)alloc((size_t)30720 * 64   * 2);
    unsigned short* qkvA  = (unsigned short*)alloc((size_t)30720 * 768  * 2);
    unsigned short* ctxo  = (unsigned short*)alloc((size_t)4096  * 1920 * 2);
    unsigned short* C3    = (unsigned short*)alloc((size_t)4096  * 1024 * 2);
    float*          C4    = (float*)alloc((size_t)1024 * 1088 * 4);

    const int TPB = 256;
    auto blocks = [](size_t n, int tpb) { return (unsigned)((n + tpb - 1) / tpb); };

    // ---- 1. pack weights (fp32 -> transposed padded bf16) ----
    pack_wt<<<blocks((size_t)256  * 160,  TPB), TPB, 0, stream>>>(iw1, iw1t, 136, 256,  160,  256);
    pack_wt<<<blocks((size_t)64   * 1024, TPB), TPB, 0, stream>>>(iw2, iw2t, 1024, 64,  1024, 64);
    pack_wt<<<blocks((size_t)1024 * 1920, TPB), TPB, 0, stream>>>(ow1, ow1t, 1920, 1024, 1920, 1024);
    pack_wt<<<blocks((size_t)1088 * 4096, TPB), TPB, 0, stream>>>(ow2, ow2t, 4096, 1080, 4096, 1088);
    pack_qkv<<<blocks((size_t)768 * 64, TPB), TPB, 0, stream>>>(owq, owk, owv, obq, obk, obv, qkvWt, qkvB);
    pad_bias<<<blocks(1088, TPB), TPB, 0, stream>>>(ob2, ob2p, 1080, 1088);

    // ---- 2. inner attention -> bf16 A for inner fc1 ----
    inner_attn<<<30720, 128, 0, stream>>>(x, iwq, ibq, iwk, ibk, iwv, ibv, ctxiA);

    // ---- 3. inner fc1 + tanh : [122880,160] @ [160,256] -> bf16 C1 ----
    gemm_bf16_wmma<<<dim3(4, 1920), 128, 0, stream>>>(ctxiA, iw1t, ib1, C1,
                                                      122880, 256, 160, 1);
    // ---- 4. inner fc2 : [30720,1024] @ [1024,64] -> bf16 seq ----
    gemm_bf16_wmma<<<dim3(1, 480), 128, 0, stream>>>(C1, iw2t, ib2, seqA,
                                                     30720, 64, 1024, 2);
    // ---- 5. outer q/k/v : [30720,64] @ [64,768] -> bf16 qkv ----
    gemm_bf16_wmma<<<dim3(12, 480), 128, 0, stream>>>(seqA, qkvWt, qkvB, qkvA,
                                                      30720, 768, 64, 2);
    // ---- 6. outer attention -> bf16 A for outer fc1 ----
    outer_attn<<<4096, 256, 0, stream>>>(qkvA, ctxo);

    // ---- 7. outer fc1 + tanh : [4096,1920] @ [1920,1024] -> bf16 C3 ----
    gemm_bf16_wmma<<<dim3(16, 64), 128, 0, stream>>>(ctxo, ow1t, ob1, C3,
                                                     4096, 1024, 1920, 1);
    // ---- 8. outer fc2 : [1024,4096] @ [4096,1088] -> f32 C4 ----
    gemm_bf16_wmma<<<dim3(17, 16), 128, 0, stream>>>(C3, ow2t, ob2p, C4,
                                                     1024, 1088, 4096, 0);
    // ---- 9. strip padding into d_out ----
    final_copy<<<blocks((size_t)1024 * 1080, TPB), TPB, 0, stream>>>(C4, (float*)d_out);
}
